// MambaVideoEncoder_31963146616889
// MI455X (gfx1250) — matrix operand order repaired and checked by reading, not compiled, not run on hardware
//
#include <hip/hip_runtime.h>

#define L_SEQ   4096
#define DMODEL  768
#define DINNER  1536
#define DSTATE  16
#define DTRANK  48
#define DTRPAD  64
#define XDBL    80   // dt_rank + 2*d_state
#define XDBL_PAD 128 // padded N for x_proj weights (8 tiles of 16)

typedef __attribute__((ext_vector_type(16))) __bf16 v16bf;
typedef __attribute__((ext_vector_type(8)))  float  v8f;

// ------------- f32 -> bf16 conversion with K padding and row padding -------------
__global__ void cvt_bf16_pad(const float* __restrict__ in, __bf16* __restrict__ out,
                             int rows_out, int rows_in, int kin, int kpad, int in_stride) {
    int idx = blockIdx.x * blockDim.x + threadIdx.x;
    int total = rows_out * kpad;
    if (idx >= total) return;
    int r = idx / kpad;
    int k = idx - r * kpad;
    float v = (k < kin && r < rows_in) ? in[(size_t)r * in_stride + k] : 0.0f;
    out[idx] = (__bf16)v;
}

// ---- WMMA bf16 GEMM, 4-way N register blocking: C(MxN,f32) = A(MxK) * W(NpadxK)^T ----
// Each wave computes a 16x64 strip: one A fragment amortizes 4 WMMAs per K-chunk.
// W must have >= 4*gridDim.x*16 rows (zero-padded); stores guarded by N.
__global__ void gemm_wmma_bf16_x4(const __bf16* __restrict__ A, const __bf16* __restrict__ W,
                                  float* __restrict__ C, int K, int N, int ldc) {
    int lane = threadIdx.x & 31;
    int wave = threadIdx.x >> 5;
    int lh   = lane >> 4;        // lane half: 0 or 1
    int l16  = lane & 15;
    int mbase = (blockIdx.y * 8 + wave) * 16;
    int nt0   = blockIdx.x * 4;  // first of 4 N tiles

    const __bf16* arow = A + (size_t)(mbase + l16) * K;        // A: M = lane&15
    const __bf16* br0  = W + (size_t)(nt0 * 16 + l16) * K;     // B: N = lane&15
    const __bf16* br1  = br0 + (size_t)16 * K;
    const __bf16* br2  = br0 + (size_t)32 * K;
    const __bf16* br3  = br0 + (size_t)48 * K;
    v8f c0 = {}, c1 = {}, c2 = {}, c3 = {};

    for (int kb = 0; kb < K; kb += 32) {
        v16bf a, b0, b1, b2, b3;
        // A 16x32 (same k order as B): lanes<16 hold K 0..15 (elts 0..15);
        // lanes>=16 hold K 16..31.
        #pragma unroll
        for (int j = 0; j < 8; ++j) a[j]     = arow[kb + lh * 16 + j];
        #pragma unroll
        for (int j = 0; j < 8; ++j) a[8 + j] = arow[kb + lh * 16 + 8 + j];
        // B 32x16: lanes<16 hold K 0..15, lanes>=16 hold K 16..31; N = lane&15.
        #pragma unroll
        for (int j = 0; j < 16; ++j) b0[j] = br0[kb + lh * 16 + j];
        #pragma unroll
        for (int j = 0; j < 16; ++j) b1[j] = br1[kb + lh * 16 + j];
        #pragma unroll
        for (int j = 0; j < 16; ++j) b2[j] = br2[kb + lh * 16 + j];
        #pragma unroll
        for (int j = 0; j < 16; ++j) b3[j] = br3[kb + lh * 16 + j];
        c0 = __builtin_amdgcn_wmma_f32_16x16x32_bf16(false, a, false, b0, (short)0, c0, false, false);
        c1 = __builtin_amdgcn_wmma_f32_16x16x32_bf16(false, a, false, b1, (short)0, c1, false, false);
        c2 = __builtin_amdgcn_wmma_f32_16x16x32_bf16(false, a, false, b2, (short)0, c2, false, false);
        c3 = __builtin_amdgcn_wmma_f32_16x16x32_bf16(false, a, false, b3, (short)0, c3, false, false);
    }
    // C/D layout: VGPR r -> M = r + 8*laneHalf, N = lane&15
    v8f acc[4] = {c0, c1, c2, c3};
    #pragma unroll
    for (int t = 0; t < 4; ++t) {
        int col = (nt0 + t) * 16 + l16;
        if (col < N) {
            #pragma unroll
            for (int r = 0; r < 8; ++r)
                C[(size_t)(mbase + r + lh * 8) * ldc + col] = acc[t][r];
        }
    }
}

// ---------------- z at the last timestep only (f32 dot) ----------------
__global__ void zlast_dot(const float* __restrict__ frame, const float* __restrict__ w_in,
                          float* __restrict__ zlast) {
    int d = blockIdx.x * blockDim.x + threadIdx.x;
    if (d >= DINNER) return;
    const float* x = frame + (size_t)(L_SEQ - 1) * DMODEL;
    const float* w = w_in + (size_t)(DINNER + d) * DMODEL;
    float acc = 0.f;
    for (int k = 0; k < DMODEL; ++k) acc += x[k] * w[k];
    zlast[d] = acc;
}

// ---------------- depthwise conv(4) + bias + SiLU ----------------
__global__ void conv_silu(const float* __restrict__ xc, const float* __restrict__ cw,
                          const float* __restrict__ cb, float* __restrict__ out) {
    int idx = blockIdx.x * blockDim.x + threadIdx.x;
    if (idx >= L_SEQ * DINNER) return;
    int l = idx / DINNER;
    int d = idx - l * DINNER;
    float acc = cb[d];
    #pragma unroll
    for (int j = 0; j < 4; ++j) {
        int ls = l - 3 + j;
        if (ls >= 0) acc += cw[d * 4 + j] * xc[(size_t)ls * DINNER + d];
    }
    out[idx] = acc / (1.0f + __expf(-acc));  // silu
}

// ---------------- softplus(dt_raw + bias), in place ----------------
__global__ void softplus_bias(float* __restrict__ dt, const float* __restrict__ bias) {
    int idx = blockIdx.x * blockDim.x + threadIdx.x;
    if (idx >= L_SEQ * DINNER) return;
    int d = idx % DINNER;
    float v = dt[idx] + bias[d];
    dt[idx] = (v > 20.0f) ? v : log1pf(__expf(v));
}

// ---------------- per-channel inclusive prefix sum over l ----------------
__global__ void prefix_dt(const float* __restrict__ dt, float* __restrict__ cum) {
    int d = blockIdx.x * blockDim.x + threadIdx.x;
    if (d >= DINNER) return;
    float c = 0.f;
    #pragma unroll 4
    for (int l = 0; l < L_SEQ; ++l) {
        c += dt[(size_t)l * DINNER + d];
        cum[(size_t)l * DINNER + d] = c;
    }
}

// ---- closed-form final state: y[d] = sum_l dt*x*sum_s B[l,s]*C_L[s]*exp(A*(T-cum)) ----
__global__ void scan_y(const float* __restrict__ dt, const float* __restrict__ cum,
                       const float* __restrict__ xc, const float* __restrict__ xdbl,
                       const float* __restrict__ A_log, const float* __restrict__ Dp,
                       const float* __restrict__ zlast, float* __restrict__ y) {
    int d = blockIdx.x;
    int tid = threadIdx.x;
    float Ar[DSTATE], Cl[DSTATE];
    #pragma unroll
    for (int s = 0; s < DSTATE; ++s) {
        Ar[s] = -__expf(A_log[d * DSTATE + s]);
        Cl[s] = xdbl[(size_t)(L_SEQ - 1) * XDBL + DTRANK + DSTATE + s];
    }
    float T = cum[(size_t)(L_SEQ - 1) * DINNER + d];
    float acc = 0.f;
    for (int l = tid; l < L_SEQ; l += 256) {
        float r = T - cum[(size_t)l * DINNER + d];           // >= 0 (A < 0 => stable)
        float a = dt[(size_t)l * DINNER + d] * xc[(size_t)l * DINNER + d];
        const float* Bp = xdbl + (size_t)l * XDBL + DTRANK;
        float t = 0.f;
        #pragma unroll
        for (int s = 0; s < DSTATE; ++s) t += Bp[s] * Cl[s] * __expf(Ar[s] * r);
        acc += a * t;
    }
    __shared__ float red[256];
    red[tid] = acc;
    __syncthreads();
    for (int o = 128; o > 0; o >>= 1) {
        if (tid < o) red[tid] += red[tid + o];
        __syncthreads();
    }
    if (tid == 0) {
        float yv = red[0] + xc[(size_t)(L_SEQ - 1) * DINNER + d] * Dp[d];
        float z  = zlast[d];
        y[d] = yv * (z / (1.0f + __expf(-z)));
    }
}

// ---------------- out_proj (768x1536) + L2 normalize, single block ----------------
__global__ void out_norm(const float* __restrict__ wout, const float* __restrict__ yv,
                         float* __restrict__ out) {
    __shared__ float ys[DINNER];
    __shared__ float sq[1024];
    int e = threadIdx.x;                        // 768 threads
    for (int d = e; d < DINNER; d += DMODEL) ys[d] = yv[d];
    __syncthreads();
    float acc = 0.f;
    const float* w = wout + (size_t)e * DINNER;
    for (int d = 0; d < DINNER; ++d) acc += w[d] * ys[d];
    sq[e] = acc * acc;
    if (e < 1024 - DMODEL) sq[DMODEL + e] = 0.f;
    __syncthreads();
    for (int o = 512; o > 0; o >>= 1) {
        if (e < o) sq[e] += sq[e + o];
        __syncthreads();
    }
    float norm = fmaxf(sqrtf(sq[0]), 1e-12f);
    out[e] = acc / norm;
}

extern "C" void kernel_launch(void* const* d_in, const int* in_sizes, int n_in,
                              void* d_out, int out_size, void* d_ws, size_t ws_size,
                              hipStream_t stream) {
    const float* frame   = (const float*)d_in[0];  // 4096x768
    const float* w_in    = (const float*)d_in[1];  // 3072x768
    const float* conv_w  = (const float*)d_in[2];  // 1536x1x4
    const float* conv_b  = (const float*)d_in[3];  // 1536
    const float* w_xp    = (const float*)d_in[4];  // 80x1536
    const float* w_dt    = (const float*)d_in[5];  // 1536x48
    const float* b_dt    = (const float*)d_in[6];  // 1536
    const float* A_log   = (const float*)d_in[7];  // 1536x16
    const float* Dp      = (const float*)d_in[8];  // 1536
    const float* w_out   = (const float*)d_in[9];  // 768x1536

    char* ws = (char*)d_ws;
    size_t off = 0;
    auto alloc = [&](size_t bytes) { char* p = ws + off; off += (bytes + 255) & ~(size_t)255; return p; };
    __bf16* bfA    = (__bf16*)alloc((size_t)L_SEQ * DMODEL * 2);    // frame bf16
    __bf16* bfWin  = (__bf16*)alloc((size_t)DINNER * DMODEL * 2);   // xc half of in_proj_w
    float*  xc     = (float*) alloc((size_t)L_SEQ * DINNER * 4);    // in_proj (xc) out
    float*  conv   = (float*) alloc((size_t)L_SEQ * DINNER * 4);    // conv+silu out
    __bf16* convbf = (__bf16*)alloc((size_t)L_SEQ * DINNER * 2);
    __bf16* bfWxp  = (__bf16*)alloc((size_t)XDBL_PAD * DINNER * 2); // row-padded to 128
    float*  xdbl   = (float*) alloc((size_t)L_SEQ * XDBL * 4);
    __bf16* bfdtr  = (__bf16*)alloc((size_t)L_SEQ * DTRPAD * 2);
    __bf16* bfWdt  = (__bf16*)alloc((size_t)DINNER * DTRPAD * 2);
    float*  dtbuf  = (float*) alloc((size_t)L_SEQ * DINNER * 4);
    float*  cumdt  = (float*) alloc((size_t)L_SEQ * DINNER * 4);
    float*  zlast  = (float*) alloc((size_t)DINNER * 4);
    float*  yvec   = (float*) alloc((size_t)DINNER * 4);
    (void)ws_size; (void)in_sizes; (void)n_in; (void)out_size;

    const int TPB = 256;
    auto blocks = [](size_t n, int t) { return (unsigned)((n + t - 1) / t); };

    // 1-2: bf16 conversions for in_proj GEMM
    cvt_bf16_pad<<<blocks((size_t)L_SEQ * DMODEL, TPB), TPB, 0, stream>>>(frame, bfA, L_SEQ, L_SEQ, DMODEL, DMODEL, DMODEL);
    cvt_bf16_pad<<<blocks((size_t)DINNER * DMODEL, TPB), TPB, 0, stream>>>(w_in, bfWin, DINNER, DINNER, DMODEL, DMODEL, DMODEL);
    // 3: xc = frame @ in_proj_w[0:1536]^T   (WMMA, 16x64 per wave)
    gemm_wmma_bf16_x4<<<dim3(DINNER / 64, L_SEQ / 128), TPB, 0, stream>>>(bfA, bfWin, xc, DMODEL, DINNER, DINNER);
    // 4: z at last timestep only
    zlast_dot<<<blocks(DINNER, TPB), TPB, 0, stream>>>(frame, w_in, zlast);
    // 5: depthwise conv + silu
    conv_silu<<<blocks((size_t)L_SEQ * DINNER, TPB), TPB, 0, stream>>>(xc, conv_w, conv_b, conv);
    // 6-7: bf16 for x_proj GEMM (weights row-padded 80 -> 128 with zeros)
    cvt_bf16_pad<<<blocks((size_t)L_SEQ * DINNER, TPB), TPB, 0, stream>>>(conv, convbf, L_SEQ, L_SEQ, DINNER, DINNER, DINNER);
    cvt_bf16_pad<<<blocks((size_t)XDBL_PAD * DINNER, TPB), TPB, 0, stream>>>(w_xp, bfWxp, XDBL_PAD, XDBL, DINNER, DINNER, DINNER);
    // 8: x_dbl = conv @ x_proj_w^T (4096 x 80; compute 128 wide, store 80)   (WMMA)
    gemm_wmma_bf16_x4<<<dim3(XDBL_PAD / 64, L_SEQ / 128), TPB, 0, stream>>>(convbf, bfWxp, xdbl, DINNER, XDBL, XDBL);
    // 9-10: dt_r (strided from x_dbl, K 48->64 zero pad) and dt_proj_w to bf16
    cvt_bf16_pad<<<blocks((size_t)L_SEQ * DTRPAD, TPB), TPB, 0, stream>>>(xdbl, bfdtr, L_SEQ, L_SEQ, DTRANK, DTRPAD, XDBL);
    cvt_bf16_pad<<<blocks((size_t)DINNER * DTRPAD, TPB), TPB, 0, stream>>>(w_dt, bfWdt, DINNER, DINNER, DTRANK, DTRPAD, DTRANK);
    // 11: dt_raw = dt_r @ dt_proj_w^T (4096 x 1536)   (WMMA)
    gemm_wmma_bf16_x4<<<dim3(DINNER / 64, L_SEQ / 128), TPB, 0, stream>>>(bfdtr, bfWdt, dtbuf, DTRPAD, DINNER, DINNER);
    // 12: softplus(dt + bias) in place
    softplus_bias<<<blocks((size_t)L_SEQ * DINNER, TPB), TPB, 0, stream>>>(dtbuf, b_dt);
    // 13: per-channel prefix sum of dt over time
    prefix_dt<<<blocks(DINNER, TPB), TPB, 0, stream>>>(dtbuf, cumdt);
    // 14: closed-form final-state reduction -> y[d]
    scan_y<<<DINNER, 256, 0, stream>>>(dtbuf, cumdt, conv, xdbl, A_log, Dp, zlast, yvec);
    // 15: out_proj + normalize
    out_norm<<<1, DMODEL, 0, stream>>>(w_out, yvec, (float*)d_out);
}
